// MyModelV1_25821343383854
// MI455X (gfx1250) — matrix operand order repaired and checked by reading, not compiled, not run on hardware
//
#include <hip/hip_runtime.h>

// Problem geometry (fixed by reference)
#define NB   32      // n*t frames
#define NC   64      // channels
#define NHW  4096    // 64*64 pixels
#define NM   512     // memory bank slots
#define NG   32      // groups
#define GEPS 1e-6f

typedef __attribute__((ext_vector_type(8)))  float  v8f;
typedef __attribute__((ext_vector_type(8)))  __bf16 bf8;
typedef __attribute__((ext_vector_type(16))) __bf16 bf16v;

__device__ __forceinline__ __bf16 f2bf(float f) {
  union { float f; unsigned u; } a; a.f = f;
  unsigned r = a.u + 0x7fffu + ((a.u >> 16) & 1u);   // round-to-nearest-even
  unsigned short hs = (unsigned short)(r >> 16);
  __bf16 b; __builtin_memcpy(&b, &hs, 2); return b;
}

// B-operand: 16 contiguous bf16 per lane (lane = N column, K contiguous)
__device__ __forceinline__ bf16v ld_b16(const __bf16* p) {
  union { bf16v v; bf8 h[2]; } u;
  u.h[0] = *(const bf8*)p; u.h[1] = *(const bf8*)(p + 8);
  return u.v;
}
// A-operand: 16 contiguous bf16 (lane = M row; kb = (lane>>4)*16)
__device__ __forceinline__ bf16v ld_a16(const __bf16* p) {
  union { bf16v v; bf8 h[2]; } u;
  u.h[0] = *(const bf8*)p; u.h[1] = *(const bf8*)(p + 8);
  return u.v;
}
__device__ __forceinline__ v8f wmma_bf16(bf16v a, bf16v b, v8f c) {
  return __builtin_amdgcn_wmma_f32_16x16x32_bf16(false, a, false, b, (short)0, c,
                                                 false, false);
}

// ---- CDNA5 async global->LDS copy (ASYNCcnt path), guarded -----------------
// Diagnostic from previous round: param 0 is
//   '__attribute__((__vector_size__(4*sizeof(int)))) int __device__ *'
// i.e. a GCC-vector int4 pointer in addrspace(1) (HIP prints AS1 as
// __device__), non-const. Mirror that exactly; LDS side in addrspace(3).
#if __has_builtin(__builtin_amdgcn_global_load_async_to_lds_b128)
#define HAVE_ASYNC_COPY 1
typedef int v4i_t __attribute__((vector_size(16)));
typedef __attribute__((address_space(1))) v4i_t* as1_v4i;
typedef __attribute__((address_space(3))) v4i_t* as3_v4i;
__device__ __forceinline__ void async_cp16(const void* g, void* l) {
  __builtin_amdgcn_global_load_async_to_lds_b128((as1_v4i)g, (as3_v4i)l, 0, 0);
}
#endif
__device__ __forceinline__ void wait_async0() {
#if __has_builtin(__builtin_amdgcn_s_wait_asynccnt)
  __builtin_amdgcn_s_wait_asynccnt(0);
#else
  asm volatile("s_wait_asynccnt 0x0" ::: "memory");
#endif
}

// Padded LDS image of the constant operands + per-wave staging.
// Pack region (first PACK_BYTES) is pre-built in ws by k_prep_pack and copied
// verbatim into LDS by each block of k_attn_wz.
struct __align__(16) SmemC {
  __bf16 mbB[NC][520];        // mb  [c][m], row stride 520 (260 dw % 64 = 4)
  __bf16 mbT[NM][72];         // mbT [m][c], row stride 72  (36 dw % 64 = 36)
  __bf16 wz[NC][72];          // wz_w[o][c]
  __bf16 ebuf[8][16][136];    // per-wave D->A staging (68 dw % 64 = 4)
};
#define PACK_BYTES (sizeof(__bf16) * (NC * 520 + NM * 72 + NC * 72))  // 149504

// ---------------- K1: GroupNorm statistics (mean, inv_std) -----------------
__global__ __launch_bounds__(256) void k_gn_stats(const float* __restrict__ x,
                                                  float* __restrict__ stats) {
  int bg = blockIdx.x;                 // [NB*NG]
  int b = bg >> 5, g = bg & 31;
  const float* xp = x + ((size_t)b * NC + (size_t)g * 2) * NHW; // 2 ch contiguous
  float s = 0.f, q = 0.f;
  for (int i = threadIdx.x; i < 2 * NHW; i += 256) {
    float v = xp[i]; s += v; q += v * v;
  }
  __shared__ float sS[256], sQ[256];
  int t = threadIdx.x;
  sS[t] = s; sQ[t] = q; __syncthreads();
  for (int off = 128; off; off >>= 1) {
    if (t < off) { sS[t] += sS[t + off]; sQ[t] += sQ[t + off]; }
    __syncthreads();
  }
  if (t == 0) {
    float inv_n = 1.0f / (2.0f * NHW);
    float mean = sS[0] * inv_n;
    float var  = sQ[0] * inv_n - mean * mean;
    stats[2 * bg]     = mean;
    stats[2 * bg + 1] = rsqrtf(var + GEPS);
  }
}

// ---------------- K2: build the padded bf16 LDS pack in ws -----------------
__global__ __launch_bounds__(256) void k_prep_pack(const float* __restrict__ mb,
                                                   const float* __restrict__ wz,
                                                   char* __restrict__ pack) {
  __bf16* mbB = (__bf16*)pack;                                    // [64][520]
  __bf16* mbT = (__bf16*)(pack + sizeof(__bf16) * NC * 520);      // [512][72]
  __bf16* wzp = (__bf16*)(pack + sizeof(__bf16) * (NC * 520 + NM * 72));
  int i = blockIdx.x * 256 + threadIdx.x;
  if (i < NC * NM) {
    int c = i >> 9, m = i & (NM - 1);
    __bf16 v = f2bf(mb[i]);
    mbB[c * 520 + m] = v;
    mbT[m * 72 + c]  = v;
  }
  if (i < NC * NC) wzp[(i >> 6) * 72 + (i & 63)] = f2bf(wz[i]);
}

// ---------------- K3: GroupNorm apply + phi = (q*C^-1/2) W^T + b -----------
// Output: phi bf16 [NB][NHW][NC]; the attention scale folded into q and bias.
__global__ __launch_bounds__(128) void k_norm_phi(
    const float* __restrict__ x, const float* __restrict__ gamma,
    const float* __restrict__ beta, const float* __restrict__ phi_w,
    const float* __restrict__ phi_b, const float* __restrict__ stats,
    __bf16* __restrict__ phi) {
  __shared__ __bf16 qT[64][72];   // [pixel][channel], padded, 16B aligned rows
  __shared__ __bf16 wS[64][72];   // phi_w [o][c] bf16
  int blk = blockIdx.x;
  int b  = blk >> 6;              // 64 pixel tiles of 64 per frame
  int p0 = (blk & 63) * 64;
  const float* xb  = x + (size_t)b * NC * NHW;
  const float* stb = stats + (size_t)b * NG * 2;
  const float scale = 0.125f;     // 1/sqrt(64)

  for (int i = threadIdx.x; i < 64 * 64; i += 128) {
    int c = i >> 6, pl = i & 63;
    float v    = xb[(size_t)c * NHW + p0 + pl];
    float mean = stb[2 * (c >> 1)], inv = stb[2 * (c >> 1) + 1];
    qT[pl][c] = f2bf(((v - mean) * inv * gamma[c] + beta[c]) * scale);
    wS[c][pl] = f2bf(phi_w[(size_t)c * 64 + pl]);
  }
  __syncthreads();

  int wave = threadIdx.x >> 5, lane = threadIdx.x & 31;
  int h = lane >> 4, ln = lane & 15;
  int kb = h * 16;
  int prow = wave * 16;

  bf16v A[2];
  A[0] = ld_a16(&qT[prow + ln][0  + kb]);
  A[1] = ld_a16(&qT[prow + ln][32 + kb]);

  for (int ot = 0; ot < 4; ot++) {
    v8f acc = {0.f, 0.f, 0.f, 0.f, 0.f, 0.f, 0.f, 0.f};
    acc = wmma_bf16(A[0], ld_b16(&wS[ot * 16 + ln][0  + h * 16]), acc);
    acc = wmma_bf16(A[1], ld_b16(&wS[ot * 16 + ln][32 + h * 16]), acc);
    float bias = phi_b[ot * 16 + ln] * scale;
    __bf16* op = phi + ((size_t)b * NHW + p0 + prow) * NC + ot * 16 + ln;
#pragma unroll
    for (int r = 0; r < 8; r++) {
      int row = r + 8 * h;
      op[(size_t)row * NC] = f2bf(acc[r] + bias);
    }
  }
}

// ---------------- K4: attention (flash softmax) + wz + residual ------------
__global__ __launch_bounds__(256) void k_attn_wz(
    const __bf16* __restrict__ phi, const char* __restrict__ pack,
    const float* __restrict__ wz_b, const float* __restrict__ x,
    float* __restrict__ out) {
  extern __shared__ char smem_raw[];
  SmemC& sm = *reinterpret_cast<SmemC*>(smem_raw);

  int blk = blockIdx.x;
  int b = blk >> 5, pt = blk & 31;      // 32 tiles of 128 px per frame
  int wave = threadIdx.x >> 5, lane = threadIdx.x & 31;
  int h = lane >> 4, ln = lane & 15;
  int kb = h * 16;
  int p0 = pt * 128 + wave * 16;

  {
    // Fill LDS with the pre-padded constant pack (mb, mbT, wz_w).
    char* dst = (char*)&sm;
    const int chunks = (int)(PACK_BYTES / 16);
#if defined(HAVE_ASYNC_COPY)
    for (int i = threadIdx.x; i < chunks; i += 256)
      async_cp16(pack + 16 * (size_t)i, dst + 16 * (size_t)i);
    wait_async0();
#else
    for (int i = threadIdx.x; i < chunks; i += 256)
      *(uint4*)(dst + 16 * (size_t)i) = *(const uint4*)(pack + 16 * (size_t)i);
#endif
  }
  __syncthreads();

  // phi A-tiles for this wave's 16 rows (K=64 -> 2 K-steps), kept in regs
  const __bf16* prow = phi + ((size_t)b * NHW + p0 + ln) * NC;
  bf16v Aphi[2];
  Aphi[0] = ld_a16(prow + 0  + kb);
  Aphi[1] = ld_a16(prow + 32 + kb);

  float mrow[8], lrow[8];
  v8f yacc[4];
#pragma unroll
  for (int r = 0; r < 8; r++) { mrow[r] = -1e30f; lrow[r] = 0.f; }
#pragma unroll
  for (int n = 0; n < 4; n++) yacc[n] = (v8f){0.f,0.f,0.f,0.f,0.f,0.f,0.f,0.f};

  for (int ch = 0; ch < 4; ch++) {        // M chunks of 128
    int m0 = ch * 128;
    v8f s[8];
#pragma unroll
    for (int nt = 0; nt < 8; nt++) {
      const __bf16* br = &sm.mbT[m0 + nt * 16 + ln][h * 16];
      v8f acc = {0.f, 0.f, 0.f, 0.f, 0.f, 0.f, 0.f, 0.f};
      acc = wmma_bf16(Aphi[0], ld_b16(br), acc);
      acc = wmma_bf16(Aphi[1], ld_b16(br + 32), acc);
      s[nt] = acc;
    }
    // online softmax: row = r + 8*h lives in VGPR r, cols across 16-lane half
    float mnew[8], fac[8];
#pragma unroll
    for (int r = 0; r < 8; r++) {
      float mx = s[0][r];
#pragma unroll
      for (int nt = 1; nt < 8; nt++) mx = fmaxf(mx, s[nt][r]);
      for (int d = 1; d < 16; d <<= 1) mx = fmaxf(mx, __shfl_xor(mx, d, 32));
      mnew[r] = fmaxf(mrow[r], mx);
      fac[r]  = __expf(mrow[r] - mnew[r]);
      mrow[r] = mnew[r];
    }
#pragma unroll
    for (int r = 0; r < 8; r++) {
      float rs = 0.f;
#pragma unroll
      for (int nt = 0; nt < 8; nt++) {
        float e = __expf(s[nt][r] - mnew[r]);
        s[nt][r] = e;
        rs += e;
      }
      for (int d = 1; d < 16; d <<= 1) rs += __shfl_xor(rs, d, 32);
      lrow[r] = lrow[r] * fac[r] + rs;
    }
#pragma unroll
    for (int n = 0; n < 4; n++)
#pragma unroll
      for (int r = 0; r < 8; r++) yacc[n][r] *= fac[r];
    // D-layout -> A-layout re-tile through per-wave LDS
#pragma unroll
    for (int nt = 0; nt < 8; nt++)
#pragma unroll
      for (int r = 0; r < 8; r++)
        sm.ebuf[wave][r + 8 * h][nt * 16 + ln] = f2bf(s[nt][r]);
    __syncthreads();
    // y1 += E(16x128) x mb^T : B[k=m][n=c] = mb[c][m] (rows of sm.mbB)
#pragma unroll
    for (int ks = 0; ks < 4; ks++) {
      bf16v Ae = ld_a16(&sm.ebuf[wave][ln][ks * 32 + kb]);
#pragma unroll
      for (int n = 0; n < 4; n++) {
        const __bf16* br = &sm.mbB[n * 16 + ln][m0 + ks * 32 + h * 16];
        yacc[n] = wmma_bf16(Ae, ld_b16(br), yacc[n]);
      }
    }
    __syncthreads();
  }

  // finalize softmax normalization
#pragma unroll
  for (int r = 0; r < 8; r++) {
    float inv = 1.f / lrow[r];
#pragma unroll
    for (int n = 0; n < 4; n++) yacc[n][r] *= inv;
  }
  // stage y1 as bf16 for the wz GEMM A-operand
#pragma unroll
  for (int n = 0; n < 4; n++)
#pragma unroll
    for (int r = 0; r < 8; r++)
      sm.ebuf[wave][r + 8 * h][n * 16 + ln] = f2bf(yacc[n][r]);
  __syncthreads();

  bf16v Ay[2];
  Ay[0] = ld_a16(&sm.ebuf[wave][ln][0  + kb]);
  Ay[1] = ld_a16(&sm.ebuf[wave][ln][32 + kb]);

  const float* xb = x + (size_t)b * NC * NHW;
  float* ob = out + (size_t)b * NC * NHW;
#pragma unroll
  for (int n = 0; n < 4; n++) {
    v8f acc = {0.f, 0.f, 0.f, 0.f, 0.f, 0.f, 0.f, 0.f};
    acc = wmma_bf16(Ay[0], ld_b16(&sm.wz[n * 16 + ln][0  + h * 16]), acc);
    acc = wmma_bf16(Ay[1], ld_b16(&sm.wz[n * 16 + ln][32 + h * 16]), acc);
    float bias = wz_b[n * 16 + ln];
    int o = n * 16 + ln;
#pragma unroll
    for (int r = 0; r < 8; r++) {
      int p = p0 + r + 8 * h;
      ob[(size_t)o * NHW + p] = acc[r] + bias + xb[(size_t)o * NHW + p];
    }
  }
}

// ---------------------------------------------------------------------------
extern "C" void kernel_launch(void* const* d_in, const int* in_sizes, int n_in,
                              void* d_out, int out_size, void* d_ws,
                              size_t ws_size, hipStream_t stream) {
  const float* x     = (const float*)d_in[0];
  const float* gamma = (const float*)d_in[1];
  const float* beta  = (const float*)d_in[2];
  const float* phi_w = (const float*)d_in[3];
  const float* phi_b = (const float*)d_in[4];
  const float* mb    = (const float*)d_in[5];
  const float* wz_w  = (const float*)d_in[6];
  const float* wz_b  = (const float*)d_in[7];
  float* out = (float*)d_out;

  char* ws = (char*)d_ws;
  float* stats = (float*)ws;                          // 8 KB
  char*  pack  = ws + 8192;                           // 149504 B padded pack
  __bf16* phi  = (__bf16*)(ws + 8192 + PACK_BYTES);   // 16 MB

  k_gn_stats<<<NB * NG, 256, 0, stream>>>(x, stats);
  k_prep_pack<<<(NC * NM + 255) / 256, 256, 0, stream>>>(mb, wz_w, pack);
  k_norm_phi<<<NB * (NHW / 64), 128, 0, stream>>>(x, gamma, beta, phi_w, phi_b,
                                                  stats, phi);
  k_attn_wz<<<NB * (NHW / 128), 256, (int)sizeof(SmemC), stream>>>(
      phi, pack, wz_b, x, out);
}
